// CausalGAT_81475529605237
// MI455X (gfx1250) — compile-verified, matrix operand order checked by reading, not run on hardware
//
#include <hip/hip_runtime.h>

// ---------------------------------------------------------------- constants
constexpr int Nn    = 20000;
constexpr int Ee    = 320000;
constexpr int EXn   = Ee + Nn;       // edges + self loops
constexpr int Bb    = 100;
constexpr int HIDc  = 128;
constexpr int HEADS = 4;
constexpr int HCc   = 32;
constexpr int NCLS  = 10;
constexpr float EPSf = 1e-5f;
constexpr float NEGs = 0.2f;

typedef __attribute__((ext_vector_type(16))) __bf16 v16bf;
typedef __attribute__((ext_vector_type(8)))  float  v8f;
typedef __attribute__((ext_vector_type(4)))  float  f32x4;

__device__ __forceinline__ void atomicMaxF(float* a, float v) {
  if (v >= 0.f) atomicMax((int*)a, __float_as_int(v));
  else          atomicMin((unsigned int*)a, __float_as_uint(v));
}

// ---------------------------------------------------------------- utility
__global__ void k_fill(float* p, float v, int n) {
  int i = blockIdx.x * blockDim.x + threadIdx.x;
  if (i < n) p[i] = v;
}
__global__ void k_copy(const float* s, float* d, int n) {
  int i = blockIdx.x * blockDim.x + threadIdx.x;
  if (i < n) d[i] = s[i];
}
__global__ void k_relu_copy(const float* s, float* d, int n) {
  int i = blockIdx.x * blockDim.x + threadIdx.x;
  if (i < n) d[i] = fmaxf(s[i], 0.f);
}

// ---------------------------------------------------------------- layernorm (one block per row)
__global__ void k_layernorm(const float* __restrict__ in, const float* __restrict__ w,
                            const float* __restrict__ b, float* __restrict__ out, int cols) {
  __shared__ float s1[256], s2[256];
  int row = blockIdx.x, c = threadIdx.x;
  float v = in[row * cols + c];
  s1[c] = v; s2[c] = v * v;
  __syncthreads();
  for (int o = blockDim.x >> 1; o > 0; o >>= 1) {
    if (c < o) { s1[c] += s1[c + o]; s2[c] += s2[c + o]; }
    __syncthreads();
  }
  float m   = s1[0] / (float)cols;
  float var = s2[0] / (float)cols - m * m;
  out[row * cols + c] = (v - m) * rsqrtf(var + EPSf) * w[c] + b[c];
}

// ---------------------------------------------------------------- batchnorm (column stats over rows)
__global__ void k_colstats(const float* __restrict__ in, float* __restrict__ stats,
                           int rows, int cols, int rows_per_blk) {
  int c  = threadIdx.x;                       // cols threads
  int r0 = blockIdx.x * rows_per_blk;
  int r1 = r0 + rows_per_blk; if (r1 > rows) r1 = rows;
  float s = 0.f, q = 0.f;
  for (int r = r0; r < r1; ++r) { float v = in[r * cols + c]; s += v; q += v * v; }
  atomicAdd(&stats[c], s);
  atomicAdd(&stats[cols + c], q);
}
__global__ void k_bn_apply(const float* __restrict__ in, const float* __restrict__ stats,
                           const float* __restrict__ w, const float* __restrict__ b,
                           float* __restrict__ out, int rows, int cols) {
  int i = blockIdx.x * blockDim.x + threadIdx.x;
  if (i >= rows * cols) return;
  int c = i % cols;
  float m   = stats[c] / (float)rows;
  float var = stats[cols + c] / (float)rows - m * m;
  out[i] = (in[i] - m) * rsqrtf(var + EPSf) * w[c] + b[c];
}

// ---------------------------------------------------------------- W pre-pack into per-lane B-fragment layout
// Bp[((tn*(K/32) + kc)*32 + lane)*16 + j]  (bf16), so each lane's fragment is one
// contiguous 32-byte block.  Fragment element j (16-bit 32x16 B layout, wave32):
//   j<8 : K = kc*32 + 8*half + j       j>=8 : K = kc*32 + 16 + 8*half + (j-8)
//   N   = tn*16 + (lane&15)
__global__ void k_pack_w(const float* __restrict__ W, __bf16* __restrict__ Bp, int K) {
  int i = blockIdx.x * blockDim.x + threadIdx.x;   // total K*128
  if (i >= K * HIDc) return;
  int j    = i & 15;
  int lane = (i >> 4) & 31;
  int rest = i >> 9;
  int kcN  = K >> 5;
  int kc   = rest % kcN;
  int tn   = rest / kcN;
  int half = lane >> 4, l = lane & 15;
  int kk = kc * 32 + (half << 3) + ((j < 8) ? j : (16 + (j - 8)));
  Bp[i] = (__bf16)W[kk * HIDc + tn * 16 + l];
}

// ---------------------------------------------------------------- WMMA bf16 GEMM: C = [relu](A @ W + bias)
// A: M x K (f32 row-major), Ncols fixed = 128.  One wave per 16-row panel; the wave
// holds 8 f32 16x16 accumulators (one per 16-col tile) and reuses each A fragment
// across 8 v_wmma_f32_16x16x32_bf16 issues per K-chunk.
__global__ void k_gemm_wmma(const float* __restrict__ A, const __bf16* __restrict__ Bp,
                            const float* __restrict__ bias, float* __restrict__ C,
                            int M, int K, int fuse_relu) {
  int wave   = blockIdx.x * blockDim.y + threadIdx.y;
  int tilesM = (M + 15) >> 4;
  if (wave >= tilesM) return;                 // wave-uniform exit
  int tm = wave;
  int lane = threadIdx.x;
  int half = lane >> 4, l = lane & 15;
  int rowA = tm * 16 + l;
  bool rowOK = rowA < M;
  int rowC = rowOK ? rowA : (M - 1);          // clamped, branch-free loads

  int kcN = K >> 5;
  v8f acc[8];
#pragma unroll
  for (int t = 0; t < 8; ++t) acc[t] = (v8f){};

  const float* Arow = A + (size_t)rowC * K;
  for (int kc = 0; kc < kcN; ++kc) {
    int k0 = kc << 5;
    // A fragment: two contiguous 8-float runs -> four b128 loads
    f32x4 a0 = *(const f32x4*)(Arow + k0 + (half << 3));
    f32x4 a1 = *(const f32x4*)(Arow + k0 + (half << 3) + 4);
    f32x4 a2 = *(const f32x4*)(Arow + k0 + 16 + (half << 3));
    f32x4 a3 = *(const f32x4*)(Arow + k0 + 16 + (half << 3) + 4);
    if (!rowOK) { a0 = (f32x4){}; a1 = (f32x4){}; a2 = (f32x4){}; a3 = (f32x4){}; }
    v16bf af;
#pragma unroll
    for (int j = 0; j < 4; ++j) {
      af[j]      = (__bf16)a0[j];
      af[4 + j]  = (__bf16)a1[j];
      af[8 + j]  = (__bf16)a2[j];
      af[12 + j] = (__bf16)a3[j];
    }
#pragma unroll
    for (int tn = 0; tn < 8; ++tn) {
      v16bf bf = *(const v16bf*)(Bp + ((((size_t)tn * kcN + kc) * 32 + lane) << 4));
      acc[tn] = __builtin_amdgcn_wmma_f32_16x16x32_bf16(false, af, false, bf,
                                                        (short)0, acc[tn], false, false);
    }
  }
#pragma unroll
  for (int tn = 0; tn < 8; ++tn) {
#pragma unroll
    for (int r = 0; r < 8; ++r) {
      int m = tm * 16 + r + half * 8;         // C layout: N = lane&15, M = r + 8*half
      if (m < M) {
        float v = acc[tn][r];
        if (bias) v += bias[tn * 16 + l];
        if (fuse_relu) v = fmaxf(v, 0.f);
        C[(size_t)m * HIDc + tn * 16 + l] = v;
      }
    }
  }
}

// ---------------------------------------------------------------- GAT pieces
__global__ void k_headdot(const float* __restrict__ xw, const float* __restrict__ aS,
                          const float* __restrict__ aD, float* __restrict__ asrc,
                          float* __restrict__ adst) {
  int i = blockIdx.x * blockDim.x + threadIdx.x;   // Nn*HEADS
  if (i >= Nn * HEADS) return;
  int h = i & 3, n = i >> 2;
  float s = 0.f, d = 0.f;
  const float* xp = xw + n * HIDc + h * HCc;
  for (int c = 0; c < HCc; ++c) { float v = xp[c]; s += v * aS[h * HCc + c]; d += v * aD[h * HCc + c]; }
  asrc[i] = s; adst[i] = d;
}
__global__ void k_edge_score(const int* __restrict__ row, const int* __restrict__ col,
                             const float* __restrict__ asrc, const float* __restrict__ adst,
                             float* __restrict__ e, float* __restrict__ mseg) {
  int i = blockIdx.x * blockDim.x + threadIdx.x;   // EXn*HEADS
  if (i >= EXn * HEADS) return;
  int eid = i >> 2, h = i & 3;
  int s, d;
  if (eid < Ee) { s = row[eid]; d = col[eid]; } else { s = d = eid - Ee; }
  float v = asrc[s * HEADS + h] + adst[d * HEADS + h];
  v = v > 0.f ? v : NEGs * v;                      // leaky relu
  e[i] = v;
  atomicMaxF(&mseg[d * HEADS + h], v);
}
__global__ void k_edge_exp(const int* __restrict__ row, const int* __restrict__ col,
                           float* __restrict__ e, const float* __restrict__ mseg,
                           float* __restrict__ sseg) {
  int i = blockIdx.x * blockDim.x + threadIdx.x;
  if (i >= EXn * HEADS) return;
  int eid = i >> 2, h = i & 3;
  int d = (eid < Ee) ? col[eid] : (eid - Ee);
  float w = expf(e[i] - mseg[d * HEADS + h]);
  e[i] = w;
  atomicAdd(&sseg[d * HEADS + h], w);
}
__global__ void k_edge_agg(const int* __restrict__ row, const int* __restrict__ col,
                           const float* __restrict__ e, const float* __restrict__ sseg,
                           const float* __restrict__ xw, float* __restrict__ agg) {
  int i = blockIdx.x * blockDim.x + threadIdx.x;   // EXn*HIDc
  if (i >= EXn * HIDc) return;
  int eid = i >> 7, c = i & 127, h = c >> 5;
  int s, d;
  if (eid < Ee) { s = row[eid]; d = col[eid]; } else { s = d = eid - Ee; }
  float alpha = e[eid * HEADS + h] / sseg[d * HEADS + h];
  atomicAdd(&agg[d * HIDc + c], alpha * xw[s * HIDc + c]);
}
__global__ void k_bias_relu(const float* __restrict__ agg, const float* __restrict__ bias,
                            float* __restrict__ h) {
  int i = blockIdx.x * blockDim.x + threadIdx.x;
  if (i >= Nn * HIDc) return;
  h[i] = fmaxf(agg[i] + bias[i & 127], 0.f);
}

// ---------------------------------------------------------------- edge / node attention
__global__ void k_eatt(const float* __restrict__ h, const int* __restrict__ row,
                       const int* __restrict__ col, const float* __restrict__ W,
                       const float* __restrict__ b, float* __restrict__ ewc,
                       float* __restrict__ ewo) {
  int eid = blockIdx.x * blockDim.x + threadIdx.x;
  if (eid >= Ee) return;
  int r = row[eid], c = col[eid];
  float z0 = b[0], z1 = b[1];
  for (int j = 0; j < HIDc; ++j) {
    float v = h[r * HIDc + j];
    z0 += v * W[j * 2 + 0]; z1 += v * W[j * 2 + 1];
  }
  for (int j = 0; j < HIDc; ++j) {
    float v = h[c * HIDc + j];
    z0 += v * W[(HIDc + j) * 2 + 0]; z1 += v * W[(HIDc + j) * 2 + 1];
  }
  float m = fmaxf(z0, z1);
  float e0 = expf(z0 - m), e1 = expf(z1 - m), inv = 1.f / (e0 + e1);
  ewc[eid] = e0 * inv; ewo[eid] = e1 * inv;
}
__global__ void k_natt_score(const float* __restrict__ h, const float* __restrict__ W,
                             const float* __restrict__ b, float* __restrict__ att) {
  int n = blockIdx.x * blockDim.x + threadIdx.x;
  if (n >= Nn) return;
  float a0 = b[0], a1 = b[1];
  for (int j = 0; j < HIDc; ++j) {
    float v = h[n * HIDc + j];
    a0 += v * W[j * 2 + 0]; a1 += v * W[j * 2 + 1];
  }
  att[n * 2 + 0] = a0; att[n * 2 + 1] = a1;
}
__global__ void k_natt_kp(float* __restrict__ att, const int* __restrict__ keyp) {
  int i = blockIdx.x * blockDim.x + threadIdx.x;
  if (i >= Bb) return;
  int n = keyp[i];
  att[n * 2 + 0] = -1000000.f;
  att[n * 2 + 1] =  1000000.f;
}
__global__ void k_split(const float* __restrict__ h, const float* __restrict__ att,
                        float* __restrict__ xc, float* __restrict__ xo) {
  int i = blockIdx.x * blockDim.x + threadIdx.x;
  if (i >= Nn * HIDc) return;
  int n = i >> 7;
  float a0 = att[n * 2], a1 = att[n * 2 + 1];
  float m = fmaxf(a0, a1);
  float e0 = expf(a0 - m), e1 = expf(a1 - m), inv = 1.f / (e0 + e1);
  float v = h[i];
  xc[i] = e0 * inv * v; xo[i] = e1 * inv * v;
}

// ---------------------------------------------------------------- GCN pieces
__global__ void k_deg(const int* __restrict__ row, const float* __restrict__ ew,
                      float* __restrict__ deg) {
  int e = blockIdx.x * blockDim.x + threadIdx.x;
  if (e >= Ee) return;
  atomicAdd(&deg[row[e]], ew[e]);
}
__global__ void k_dinv(float* __restrict__ deg) {
  int n = blockIdx.x * blockDim.x + threadIdx.x;
  if (n >= Nn) return;
  float d = deg[n];
  deg[n] = d > 0.f ? rsqrtf(d) : 0.f;
}
__global__ void k_gcn_init(const float* __restrict__ xw, const float* __restrict__ dinv,
                           const float* __restrict__ bias, float* __restrict__ out) {
  int i = blockIdx.x * blockDim.x + threadIdx.x;
  if (i >= Nn * HIDc) return;
  int n = i >> 7;
  float di = dinv[n];
  out[i] = di * di * xw[i] + bias[i & 127];
}
__global__ void k_gcn_scatter(const int* __restrict__ row, const int* __restrict__ col,
                              const float* __restrict__ ew, const float* __restrict__ dinv,
                              const float* __restrict__ xw, float* __restrict__ out) {
  int i = blockIdx.x * blockDim.x + threadIdx.x;   // Ee*HIDc
  if (i >= Ee * HIDc) return;
  int e = i >> 7, c = i & 127;
  int r = row[e], cl = col[e];
  float nrm = dinv[r] * ew[e] * dinv[cl];
  atomicAdd(&out[cl * HIDc + c], nrm * xw[r * HIDc + c]);
}

// ---------------------------------------------------------------- attention pooling
__global__ void k_pool_score(const float* __restrict__ xl, const int* __restrict__ batch,
                             const int* __restrict__ keyp, float* __restrict__ score,
                             float* __restrict__ pm) {
  int n = blockIdx.x * blockDim.x + threadIdx.x;
  if (n >= Nn) return;
  int b = batch[n], kp = keyp[b];
  float s = 0.f;
  for (int j = 0; j < HIDc; ++j) s += xl[n * HIDc + j] * xl[kp * HIDc + j];
  score[n] = s;
  atomicMaxF(&pm[b], s);
}
__global__ void k_pool_exp(const float* __restrict__ score, const int* __restrict__ batch,
                           const float* __restrict__ pm, float* __restrict__ wexp,
                           float* __restrict__ ps) {
  int n = blockIdx.x * blockDim.x + threadIdx.x;
  if (n >= Nn) return;
  int b = batch[n];
  float w = expf(score[n] - pm[b]);
  wexp[n] = w;
  atomicAdd(&ps[b], w);
}
__global__ void k_pool_scatter(const float* __restrict__ xl, const int* __restrict__ batch,
                               const float* __restrict__ wexp, const float* __restrict__ ps,
                               float* __restrict__ pool) {
  int i = blockIdx.x * blockDim.x + threadIdx.x;   // Nn*HIDc
  if (i >= Nn * HIDc) return;
  int n = i >> 7, c = i & 127;
  int b = batch[n];
  atomicAdd(&pool[b * HIDc + c], xl[i] * wexp[n] / ps[b]);
}

// ---------------------------------------------------------------- small FC (N cols = NCLS)
__global__ void k_fc_small(const float* __restrict__ t, const float* __restrict__ W,
                           const float* __restrict__ b, float* __restrict__ out,
                           int rows, int K) {
  int i = blockIdx.x * blockDim.x + threadIdx.x;   // rows*NCLS
  if (i >= rows * NCLS) return;
  int r = i / NCLS, k = i % NCLS;
  float s = b[k];
  for (int j = 0; j < K; ++j) s += t[r * K + j] * W[j * NCLS + k];
  out[i] = s;
}
__global__ void k_concat(const float* __restrict__ poolo, const float* __restrict__ x,
                         const int* __restrict__ keyp, float* __restrict__ to) {
  int i = blockIdx.x * blockDim.x + threadIdx.x;   // Bb*256
  if (i >= Bb * 256) return;
  int b = i >> 8, c = i & 255;
  to[i] = (c < HIDc) ? poolo[b * HIDc + c] : x[keyp[b] * HIDc + (c - HIDc)];
}

// ================================================================ host side
static inline unsigned cdiv(long a, long b) { return (unsigned)((a + b - 1) / b); }

static void launch_gemm(const float* A, const float* W, const float* bias, float* C,
                        int M, int K, int relu, __bf16* Bp, hipStream_t st) {
  k_pack_w<<<cdiv((long)K * HIDc, 256), 256, 0, st>>>(W, Bp, K);
  int tilesM = (M + 15) >> 4;
  dim3 blk(32, 4);
  k_gemm_wmma<<<cdiv(tilesM, 4), blk, 0, st>>>(A, Bp, bias, C, M, K, relu);
}

extern "C" void kernel_launch(void* const* d_in, const int* in_sizes, int n_in,
                              void* d_out, int out_size, void* d_ws, size_t ws_size,
                              hipStream_t stream) {
  (void)in_sizes; (void)n_in; (void)out_size; (void)ws_size;
  const float* x    = (const float*)d_in[0];
  const int*   eidx = (const int*)d_in[1];
  const int*   row  = eidx;
  const int*   col  = eidx + Ee;
  const int*   batch = (const int*)d_in[2];
  const int*   keyp  = (const int*)d_in[3];
#define PI(i) ((const float*)d_in[4 + (i)])
  const float *bn_feat_w = PI(0), *bn_feat_b = PI(1);
  const float *conv_W = PI(2), *conv_b = PI(3);
  // gat block i: 4+6i .. 9+6i
  const float *eatt_W = PI(22), *eatt_b = PI(23);
  const float *natt_W = PI(24), *natt_b = PI(25);
  const float *bnc_w = PI(26), *bnc_b = PI(27), *bno_w = PI(28), *bno_b = PI(29);
  const float *cconv_W = PI(30), *cconv_b = PI(31), *oconv_W = PI(32), *oconv_b = PI(33);
  const float *pool_w = PI(34), *pool_b = PI(35);
  const float *c_ln1_w = PI(36), *c_ln1_b = PI(37), *c_fc1_W = PI(38), *c_fc1_b = PI(39);
  const float *c_ln2_w = PI(40), *c_ln2_b = PI(41), *c_fc2_W = PI(42), *c_fc2_b = PI(43);
  const float *o_ln1_w = PI(44), *o_ln1_b = PI(45), *o_fc1_W = PI(46), *o_fc1_b = PI(47);
  const float *o_ln2_w = PI(48), *o_ln2_b = PI(49), *o_fc2_W = PI(50), *o_fc2_b = PI(51);

  // workspace arena
  float* wsf = (float*)d_ws;
  size_t off = 0;
  auto A = [&](size_t n) { float* p = wsf + off; off += n; return p; };
  float* h    = A((size_t)Nn * HIDc);
  float* t0   = A((size_t)Nn * HIDc);
  float* xw   = A((size_t)Nn * HIDc);
  float* agg  = A((size_t)Nn * HIDc);
  float* xc   = A((size_t)Nn * HIDc);
  float* xo   = A((size_t)Nn * HIDc);
  float* eexp = A((size_t)EXn * HEADS);
  float* asrc = A((size_t)Nn * HEADS);
  float* adst = A((size_t)Nn * HEADS);
  float* mseg = A((size_t)Nn * HEADS);
  float* sseg = A((size_t)Nn * HEADS);
  float* ewc  = A((size_t)Ee);
  float* ewo  = A((size_t)Ee);
  float* att  = A((size_t)Nn * 2);
  float* deg  = A((size_t)Nn);
  float* stats = A(512);
  float* score = A((size_t)Nn);
  float* wexp  = A((size_t)Nn);
  float* pm    = A((size_t)Bb);
  float* ps    = A((size_t)Bb);
  float* poolc = A((size_t)Bb * HIDc);
  float* poolo = A((size_t)Bb * HIDc);
  float* tb    = A((size_t)Bb * 256);
  float* tb2   = A((size_t)Bb * 256);
  __bf16* Bp   = (__bf16*)A(16384);        // packed weights: up to 256*128 bf16

  float* outp = (float*)d_out;
  const int T = 256;

  // ---- input layernorm + conv_feat
  k_layernorm<<<Nn, HIDc, 0, stream>>>(x, bn_feat_w, bn_feat_b, t0, HIDc);
  launch_gemm(t0, conv_W, conv_b, h, Nn, HIDc, 1, Bp, stream);

  // ---- 3 GAT layers
  for (int i = 0; i < 3; ++i) {
    const float *bnw = PI(4 + 6 * i), *bnb = PI(5 + 6 * i), *gW = PI(6 + 6 * i);
    const float *aS = PI(7 + 6 * i), *aD = PI(8 + 6 * i), *gb = PI(9 + 6 * i);
    k_fill<<<cdiv(512, T), T, 0, stream>>>(stats, 0.f, 512);
    k_colstats<<<cdiv(Nn, 128), HIDc, 0, stream>>>(h, stats, Nn, HIDc, 128);
    k_bn_apply<<<cdiv((long)Nn * HIDc, T), T, 0, stream>>>(h, stats, bnw, bnb, t0, Nn, HIDc);
    launch_gemm(t0, gW, nullptr, xw, Nn, HIDc, 0, Bp, stream);
    k_headdot<<<cdiv((long)Nn * HEADS, T), T, 0, stream>>>(xw, aS, aD, asrc, adst);
    k_fill<<<cdiv((long)Nn * HEADS, T), T, 0, stream>>>(mseg, -1e30f, Nn * HEADS);
    k_edge_score<<<cdiv((long)EXn * HEADS, T), T, 0, stream>>>(row, col, asrc, adst, eexp, mseg);
    k_fill<<<cdiv((long)Nn * HEADS, T), T, 0, stream>>>(sseg, 0.f, Nn * HEADS);
    k_edge_exp<<<cdiv((long)EXn * HEADS, T), T, 0, stream>>>(row, col, eexp, mseg, sseg);
    k_fill<<<cdiv((long)Nn * HIDc, T), T, 0, stream>>>(agg, 0.f, Nn * HIDc);
    k_edge_agg<<<cdiv((long)EXn * HIDc, T), T, 0, stream>>>(row, col, eexp, sseg, xw, agg);
    k_bias_relu<<<cdiv((long)Nn * HIDc, T), T, 0, stream>>>(agg, gb, h);
  }

  // ---- edge attention & node attention split
  k_eatt<<<cdiv(Ee, 128), 128, 0, stream>>>(h, row, col, eatt_W, eatt_b, ewc, ewo);
  k_natt_score<<<cdiv(Nn, 128), 128, 0, stream>>>(h, natt_W, natt_b, att);
  k_natt_kp<<<cdiv(Bb, 128), 128, 0, stream>>>(att, keyp);
  k_split<<<cdiv((long)Nn * HIDc, T), T, 0, stream>>>(h, att, xc, xo);

  // ---- two GCN branches
  struct Branch { float* xb; const float *bw, *bbp, *W, *bias; float* ew; };
  Branch br[2] = { { xc, bnc_w, bnc_b, cconv_W, cconv_b, ewc },
                   { xo, bno_w, bno_b, oconv_W, oconv_b, ewo } };
  for (int i = 0; i < 2; ++i) {
    k_fill<<<cdiv(512, T), T, 0, stream>>>(stats, 0.f, 512);
    k_colstats<<<cdiv(Nn, 128), HIDc, 0, stream>>>(br[i].xb, stats, Nn, HIDc, 128);
    k_bn_apply<<<cdiv((long)Nn * HIDc, T), T, 0, stream>>>(br[i].xb, stats, br[i].bw, br[i].bbp, t0, Nn, HIDc);
    launch_gemm(t0, br[i].W, nullptr, xw, Nn, HIDc, 0, Bp, stream);
    k_fill<<<cdiv(Nn, T), T, 0, stream>>>(deg, 1.0f, Nn);
    k_deg<<<cdiv(Ee, T), T, 0, stream>>>(row, br[i].ew, deg);
    k_dinv<<<cdiv(Nn, T), T, 0, stream>>>(deg);
    k_gcn_init<<<cdiv((long)Nn * HIDc, T), T, 0, stream>>>(xw, deg, br[i].bias, agg);
    k_gcn_scatter<<<cdiv((long)Ee * HIDc, T), T, 0, stream>>>(row, col, br[i].ew, deg, xw, agg);
    k_relu_copy<<<cdiv((long)Nn * HIDc, T), T, 0, stream>>>(agg, br[i].xb, Nn * HIDc);
  }

  // ---- attention pooling for both branches
  float* pools[2] = { poolc, poolo };
  for (int i = 0; i < 2; ++i) {
    k_layernorm<<<Nn, HIDc, 0, stream>>>(br[i].xb, pool_w, pool_b, t0, HIDc);
    k_fill<<<cdiv(Bb, T), T, 0, stream>>>(pm, -1e30f, Bb);
    k_pool_score<<<cdiv(Nn, 128), 128, 0, stream>>>(t0, batch, keyp, score, pm);
    k_fill<<<cdiv(Bb, T), T, 0, stream>>>(ps, 0.f, Bb);
    k_pool_exp<<<cdiv(Nn, T), T, 0, stream>>>(score, batch, pm, wexp, ps);
    k_fill<<<cdiv((long)Bb * HIDc, T), T, 0, stream>>>(pools[i], 0.f, Bb * HIDc);
    k_pool_scatter<<<cdiv((long)Nn * HIDc, T), T, 0, stream>>>(t0, batch, wexp, ps, pools[i]);
  }

  // ---- causal head:  LN -> fc1(relu) -> LN -> fc2
  k_layernorm<<<Bb, HIDc, 0, stream>>>(poolc, c_ln1_w, c_ln1_b, tb, HIDc);
  launch_gemm(tb, c_fc1_W, c_fc1_b, tb2, Bb, HIDc, 1, Bp, stream);
  k_layernorm<<<Bb, HIDc, 0, stream>>>(tb2, c_ln2_w, c_ln2_b, tb, HIDc);
  k_fc_small<<<cdiv((long)Bb * NCLS, 128), 128, 0, stream>>>(tb, c_fc2_W, c_fc2_b, outp, Bb, HIDc);

  // ---- object head:  concat -> LN(256) -> fc1(relu) -> LN -> fc2
  k_concat<<<cdiv((long)Bb * 256, 128), 128, 0, stream>>>(poolo, x, keyp, tb);
  k_layernorm<<<Bb, 256, 0, stream>>>(tb, o_ln1_w, o_ln1_b, tb2, 256);
  launch_gemm(tb2, o_fc1_W, o_fc1_b, tb, Bb, 256, 1, Bp, stream);
  k_layernorm<<<Bb, HIDc, 0, stream>>>(tb, o_ln2_w, o_ln2_b, tb2, HIDc);
  k_fc_small<<<cdiv((long)Bb * NCLS, 128), 128, 0, stream>>>(tb2, o_fc2_W, o_fc2_b, outp + Bb * NCLS, Bb, HIDc);

  // ---- third output: pooled xo
  k_copy<<<cdiv((long)Bb * HIDc, T), T, 0, stream>>>(poolo, outp + 2 * Bb * NCLS, Bb * HIDc);
}
